// MultiHeadAttention_88132728914541
// MI455X (gfx1250) — matrix operand order repaired and checked by reading, not compiled, not run on hardware
//
#include <hip/hip_runtime.h>
#include <hip/hip_bf16.h>

// B=8, T=1024, D=768, H=12, DH=64
#define Bsz 8
#define Tsz 1024
#define Dsz 768
#define Hsz 12
#define DHsz 64
#define NEGV (-1e12f)

typedef __attribute__((ext_vector_type(16))) _Float16 v16h;
typedef __attribute__((ext_vector_type(8)))  float    v8f;

union AF { v16h v; _Float16 h[16]; unsigned u[8]; };

// ---------------------------------------------------------------------------
// A-fragment (16x32, f16), row-major source with leading dim `ld`.
// ISA 7.12.2: lane<16 -> M=lane, K pairs {0,1..6,7} in v0..3, {16..23} in v4..7
//             lane>=16 -> M=lane-16, K pairs {8..15} and {24..31}
// ---------------------------------------------------------------------------
__device__ __forceinline__ v16h load_a_frag(const _Float16* src, int row0, int ld,
                                            int kb, int lane) {
    int half = lane >> 4;
    int m = row0 + (lane & 15);
    AF f;
    const _Float16* rp = src + (long)m * ld + kb + 8 * half;
#pragma unroll
    for (int v = 0; v < 4; ++v) f.u[v]     = *(const unsigned*)(rp + 2 * v);
#pragma unroll
    for (int v = 0; v < 4; ++v) f.u[4 + v] = *(const unsigned*)(rp + 16 + 2 * v);
    return f.v;
}

// ---------------------------------------------------------------------------
// B-fragment (32x16, f16), row-major [K x N] source, leading dim `ld`.
// lane -> column n0+(lane&15); elem s -> K = 8*half+s (s<8), 16+8*half+(s-8)
// ---------------------------------------------------------------------------
__device__ __forceinline__ v16h load_b_frag(const _Float16* src, int kb, int ld,
                                            int n0, int lane) {
    AF f;
    int half = lane >> 4;
    const _Float16* rp = src + (long)(kb + 8 * half) * ld + n0 + (lane & 15);
#pragma unroll
    for (int s = 0; s < 8; ++s) {
        f.h[s]     = rp[(long)s * ld];
        f.h[8 + s] = rp[(long)(16 + s) * ld];
    }
    return f.v;
}

__device__ __forceinline__ v8f wmma_f16(v16h a, v16h b, v8f c) {
    return __builtin_amdgcn_wmma_f32_16x16x32_f16(false, a, false, b,
                                                  (short)0, c, false, false);
}

// ---------------------------------------------------------------------------
// f32 -> f16 conversion
// ---------------------------------------------------------------------------
__global__ void cvt_f16_kernel(const float* __restrict__ src,
                               _Float16* __restrict__ dst, int n) {
    int i = blockIdx.x * blockDim.x + threadIdx.x;
    if (i < n) dst[i] = (_Float16)src[i];
}

// ---------------------------------------------------------------------------
// QKV projection: one wave -> 16 rows x 64 cols (one head) of one of Q/K/V.
// Q,V stored [B,H,T,DH] f16; K stored transposed [B,H,DH,T] f16.
// ---------------------------------------------------------------------------
__global__ void qkv_gemm_kernel(const _Float16* __restrict__ xh,
                                const _Float16* __restrict__ Wh,
                                const float* __restrict__ bq,
                                const float* __restrict__ bk,
                                const float* __restrict__ bv,
                                _Float16* __restrict__ Qws,
                                _Float16* __restrict__ Ktws,
                                _Float16* __restrict__ Vws) {
    int wid  = (blockIdx.x * blockDim.x + threadIdx.x) >> 5;
    int lane = threadIdx.x & 31;
    int mtile = wid % (Bsz * Tsz / 16);          // 512
    int hh    = (wid / (Bsz * Tsz / 16)) % Hsz;  // 12
    int sel   = wid / ((Bsz * Tsz / 16) * Hsz);  // 0..2
    int m0 = mtile * 16;
    int n0 = hh * DHsz;
    const _Float16* W = Wh + (long)sel * Dsz * Dsz;

    v8f acc[4] = {};
    for (int kc = 0; kc < Dsz / 32; ++kc) {
        v16h a = load_a_frag(xh, m0, Dsz, kc * 32, lane);
#pragma unroll
        for (int j = 0; j < 4; ++j) {
            v16h bfr = load_b_frag(W, kc * 32, Dsz, n0 + j * 16, lane);
            acc[j] = wmma_f16(a, bfr, acc[j]);
        }
    }

    const float* bias = (sel == 0) ? bq : ((sel == 1) ? bk : bv);
    int half = lane >> 4, col = lane & 15;
#pragma unroll
    for (int j = 0; j < 4; ++j) {
        int dh = j * 16 + col;
        float bi = bias[n0 + dh];
#pragma unroll
        for (int r = 0; r < 8; ++r) {
            int m  = m0 + r + 8 * half;     // C/D layout: row = r + 8*half
            int bb = m >> 10;               // /T
            int t  = m & (Tsz - 1);
            _Float16 val = (_Float16)(acc[j][r] + bi);
            long bh = (long)bb * Hsz + hh;
            if (sel == 0)
                Qws[(bh * Tsz + t) * DHsz + dh] = val;
            else if (sel == 1)
                Ktws[(bh * DHsz + dh) * Tsz + t] = val;
            else
                Vws[(bh * Tsz + t) * DHsz + dh] = val;
        }
    }
}

// ---------------------------------------------------------------------------
// Flash attention: one wave per (b, h, 16-query tile); 32 keys per step.
// ---------------------------------------------------------------------------
__global__ void attn_kernel(const _Float16* __restrict__ Qws,
                            const _Float16* __restrict__ Ktws,
                            const _Float16* __restrict__ Vws,
                            const float* __restrict__ selector,
                            const unsigned char* __restrict__ mask,
                            float* __restrict__ out) {
    __shared__ _Float16 pbuf[8][16][32];   // per-wave P staging (C->A repack)
    int wv = threadIdx.x >> 5, lane = threadIdx.x & 31;
    int wid = blockIdx.x * 8 + wv;
    int qt = wid % (Tsz / 16);             // 64
    int hh = (wid / (Tsz / 16)) % Hsz;
    int bb = wid / ((Tsz / 16) * Hsz);
    int q0 = qt * 16;
    long bh = (long)bb * Hsz + hh;

    const _Float16* Qb  = Qws  + bh * Tsz * DHsz;
    const _Float16* Ktb = Ktws + bh * DHsz * Tsz;
    const _Float16* Vb  = Vws  + bh * Tsz * DHsz;
    const float* selb   = selector + bh * (long)Tsz * Tsz;
    const unsigned char* mb = mask + (long)bb * Tsz * Tsz;

    int half = lane >> 4, col = lane & 15;

    v16h aq0 = load_a_frag(Qb, q0, DHsz, 0, lane);
    v16h aq1 = load_a_frag(Qb, q0, DHsz, 32, lane);

    float mrun[8], lrun[8];
#pragma unroll
    for (int r = 0; r < 8; ++r) { mrun[r] = -1e30f; lrun[r] = 0.0f; }
    v8f acc[4] = {};
    const float scale = 0.125f;            // DH^-0.5

    for (int kt = 0; kt < Tsz / 32; ++kt) {
        int n0 = kt * 32;
        if (kt + 1 < Tsz / 32)             // gfx1250 global_prefetch_b8
            __builtin_prefetch(&selb[(long)(q0 + 8 * half) * Tsz + n0 + 32], 0, 1);

        // scores: two 16x16 tiles over DH=64 (two K-chunks each)
        float sv[2][8];
#pragma unroll
        for (int t = 0; t < 2; ++t) {
            v16h bk0 = load_b_frag(Ktb, 0,  Tsz, n0 + t * 16, lane);
            v16h bk1 = load_b_frag(Ktb, 32, Tsz, n0 + t * 16, lane);
            v8f c = {};
            c = wmma_f16(aq0, bk0, c);
            c = wmma_f16(aq1, bk1, c);
#pragma unroll
            for (int r = 0; r < 8; ++r) {
                int qrow = q0 + r + 8 * half;
                int kcol = n0 + t * 16 + col;
                long off = (long)qrow * Tsz + kcol;
                float sc = c[r] * scale + __logf(selb[off] + 1e-20f);
                if (!mb[off]) sc = NEGV;
                sv[t][r] = sc;
            }
        }

        // online softmax per row (rows live within 16-lane halves)
#pragma unroll
        for (int r = 0; r < 8; ++r) {
            float rm = fmaxf(sv[0][r], sv[1][r]);
            rm = fmaxf(rm, __shfl_xor(rm, 1, 32));
            rm = fmaxf(rm, __shfl_xor(rm, 2, 32));
            rm = fmaxf(rm, __shfl_xor(rm, 4, 32));
            rm = fmaxf(rm, __shfl_xor(rm, 8, 32));
            float mnew = fmaxf(mrun[r], rm);
            float corr = __expf(mrun[r] - mnew);
            float p0 = __expf(sv[0][r] - mnew);
            float p1 = __expf(sv[1][r] - mnew);
            float rs = p0 + p1;
            rs += __shfl_xor(rs, 1, 32);
            rs += __shfl_xor(rs, 2, 32);
            rs += __shfl_xor(rs, 4, 32);
            rs += __shfl_xor(rs, 8, 32);
            lrun[r] = lrun[r] * corr + rs;
            mrun[r] = mnew;
            int row = r + 8 * half;
            pbuf[wv][row][col]      = (_Float16)p0;
            pbuf[wv][row][16 + col] = (_Float16)p1;
#pragma unroll
            for (int j = 0; j < 4; ++j) acc[j][r] *= corr;
        }

        // repack P (C layout in LDS row-major) -> A fragment; accumulate P.V
        v16h ap = load_a_frag(&pbuf[wv][0][0], 0, 32, 0, lane);
#pragma unroll
        for (int j = 0; j < 4; ++j) {
            v16h bvf = load_b_frag(Vb, n0, DHsz, j * 16, lane);
            acc[j] = wmma_f16(ap, bvf, acc[j]);
        }
    }

    // epilogue: normalize and write [B,T,D] f32
    float* ob = out + (long)bb * Tsz * Dsz;
#pragma unroll
    for (int j = 0; j < 4; ++j) {
#pragma unroll
        for (int r = 0; r < 8; ++r) {
            int qrow = q0 + r + 8 * half;
            int dh = j * 16 + col;
            ob[(long)qrow * Dsz + hh * DHsz + dh] = acc[j][r] / lrun[r];
        }
    }
}

// ---------------------------------------------------------------------------
extern "C" void kernel_launch(void* const* d_in, const int* in_sizes, int n_in,
                              void* d_out, int out_size, void* d_ws, size_t ws_size,
                              hipStream_t stream) {
    const float* x  = (const float*)d_in[0];
    const float* Wq = (const float*)d_in[1];
    const float* bq = (const float*)d_in[2];
    const float* Wk = (const float*)d_in[3];
    const float* bk = (const float*)d_in[4];
    const float* Wv = (const float*)d_in[5];
    const float* bv = (const float*)d_in[6];
    const float* selector = (const float*)d_in[7];
    const unsigned char* mask = (const unsigned char*)d_in[8];
    float* out = (float*)d_out;

    const long NX  = (long)Bsz * Tsz * Dsz;   // 6,291,456
    const long ND2 = (long)Dsz * Dsz;         // 589,824

    _Float16* xh   = (_Float16*)d_ws;
    _Float16* Wh   = xh + NX;
    _Float16* Qws  = Wh + 3 * ND2;
    _Float16* Ktws = Qws + NX;
    _Float16* Vws  = Ktws + NX;

    // stage 1: f32 -> f16 conversions
    cvt_f16_kernel<<<(int)((NX + 255) / 256), 256, 0, stream>>>(x, xh, (int)NX);
    cvt_f16_kernel<<<(int)((ND2 + 255) / 256), 256, 0, stream>>>(Wq, Wh, (int)ND2);
    cvt_f16_kernel<<<(int)((ND2 + 255) / 256), 256, 0, stream>>>(Wk, Wh + ND2, (int)ND2);
    cvt_f16_kernel<<<(int)((ND2 + 255) / 256), 256, 0, stream>>>(Wv, Wh + 2 * ND2, (int)ND2);

    // stage 2: QKV projections (3 * 12 heads * 512 row-tiles waves, 8 waves/block)
    int gemm_waves = 3 * Hsz * (Bsz * Tsz / 16);     // 18432
    qkv_gemm_kernel<<<gemm_waves / 8, 256, 0, stream>>>(xh, Wh, bq, bk, bv,
                                                        Qws, Ktws, Vws);

    // stage 3: flash attention (8*12*64 = 6144 waves, 8 waves/block)
    int attn_waves = Bsz * Hsz * (Tsz / 16);         // 6144
    attn_kernel<<<attn_waves / 8, 256, 0, stream>>>(Qws, Ktws, Vws,
                                                    selector, mask, out);
}
